// ShepardsLayer_87316685127954
// MI455X (gfx1250) — matrix operand order repaired and checked by reading, not compiled, not run on hardware
//
#include <hip/hip_runtime.h>
#include <hip/hip_bf16.h>

// ---------------- problem constants ----------------
#define BATCH 2
#define SEQ   2048
#define DIM   1024
#define DQK   512
#define DVG   512
#define HEADS 8
#define DH    64                 // DQK / HEADS
#define NQ    2048               // 2*DQK + 2*DVG (qkvg width)
#define ROWS  (BATCH * SEQ)      // 4096
#define QTILES (SEQ / 16)        // 128 query tiles per (b,h)

typedef __bf16 bf16;
typedef __attribute__((ext_vector_type(16))) __bf16 v16bf;
typedef __attribute__((ext_vector_type(8)))  __bf16 v8bf;
typedef __attribute__((ext_vector_type(8)))  float  v8f;
typedef unsigned int v4u __attribute__((ext_vector_type(4)));
typedef int v4i __attribute__((ext_vector_type(4)));
typedef int v8i __attribute__((ext_vector_type(8)));

union U16B { v16bf v; v8bf h[2]; __bf16 e[16]; };

static __device__ inline v8f zero8() {
    v8f z; for (int i = 0; i < 8; ++i) z[i] = 0.0f; return z;
}

static __device__ inline v8f wmma_bf16(v16bf a, v16bf b, v8f c) {
    // D = A(16x32 bf16) * B(32x16 bf16) + C(16x16 f32)
    return __builtin_amdgcn_wmma_f32_16x16x32_bf16(
        false, a, false, b, (short)0, c, false, false);
}

// A-operand load, 16-bit A 16x32 layout:
//  lane L holds row (L&15); element e -> k = kc + (e/8)*16 + ((L>=16)?8:0) + (e%8)
static __device__ inline v16bf load_a32(const bf16* rowptr, int kc, int lane) {
    const int hi8 = (lane >= 16) ? 8 : 0;
    U16B u;
    u.h[0] = *(const v8bf*)(rowptr + kc + hi8);
    u.h[1] = *(const v8bf*)(rowptr + kc + hi8 + 16);
    return u.v;
}

// TDM: DMA a 16-row x 128-byte tile (row stride 4096 B) from global into LDS.
// D# group0: count=1 | lds_addr | 57-bit global addr | type=2 ("image")
// D# group1: data_size=8B, tensor 16x16 units, tile 16x16, dim0 stride = 512 units
static __device__ inline void tdm_load_k_tile(unsigned int lds_addr,
                                              unsigned int ga_lo, unsigned int ga_hi) {
    v4u g0;
    g0[0] = 1u;                                        // count = 1 (valid descriptor)
    g0[1] = lds_addr;                                  // LDS byte address
    g0[2] = ga_lo;                                     // global_addr[31:0]
    g0[3] = (ga_hi & 0x01FFFFFFu) | (2u << 30);        // global_addr[56:32] | type=2
    v8i g1;
    g1[0] = (int)(3u << 16);                           // data_size = 3 (8 bytes)
    g1[1] = (int)(16u << 16);                          // tensor_dim0 = 16 units (128 B row)
    g1[2] = (int)(16u << 16);                          // tensor_dim1 = 16 rows
    g1[3] = (int)(16u << 16);                          // tile_dim0 = 16 units
    g1[4] = 16;                                        // tile_dim1 = 16 rows, tile_dim2 = 0
    g1[5] = 512;                                       // tensor_dim0_stride = 4096 B / 8
    g1[6] = 0;
    g1[7] = 0;
    v4i gz; gz[0] = 0; gz[1] = 0; gz[2] = 0; gz[3] = 0;
#if __clang_major__ >= 23
    v8i gz8; for (int i = 0; i < 8; ++i) gz8[i] = 0;
    __builtin_amdgcn_tensor_load_to_lds(g0, g1, gz, gz, gz8, 0);
#else
    __builtin_amdgcn_tensor_load_to_lds(g0, g1, gz, gz, 0);
#endif
}

// ---------------- kernel 1: LayerNorm -> bf16 ----------------
__global__ __launch_bounds__(256)
void k_layernorm(const float* __restrict__ X, const float* __restrict__ gamma,
                 const float* __restrict__ beta, bf16* __restrict__ Xn) {
    const int row = blockIdx.x;
    const float* x = X + (size_t)row * DIM;
    float s = 0.0f, s2 = 0.0f;
    for (int i = threadIdx.x; i < DIM; i += 256) {
        float v = x[i]; s += v; s2 += v * v;
    }
    for (int o = 16; o > 0; o >>= 1) {
        s  += __shfl_down(s,  o, 32);
        s2 += __shfl_down(s2, o, 32);
    }
    __shared__ float sbuf[8], s2buf[8];
    const int wid = threadIdx.x >> 5, lane = threadIdx.x & 31;
    if (lane == 0) { sbuf[wid] = s; s2buf[wid] = s2; }
    __syncthreads();
    __shared__ float smu, srstd;
    if (threadIdx.x == 0) {
        float ts = 0.0f, ts2 = 0.0f;
        for (int i = 0; i < 8; ++i) { ts += sbuf[i]; ts2 += s2buf[i]; }
        float mu  = ts / (float)DIM;
        float var = ts2 / (float)DIM - mu * mu;
        smu = mu; srstd = __frsqrt_rn(var + 1e-5f);
    }
    __syncthreads();
    const float mu = smu, rstd = srstd;
    for (int i = threadIdx.x; i < DIM; i += 256) {
        float v = (x[i] - mu) * rstd * gamma[i] + beta[i];
        Xn[(size_t)row * DIM + i] = (bf16)v;
    }
}

// ---------------- kernel 2: f32 -> bf16 convert ----------------
__global__ __launch_bounds__(256)
void k_cvt_bf16(const float* __restrict__ src, bf16* __restrict__ dst, int n) {
    int i = blockIdx.x * 256 + threadIdx.x;
    if (i < n) dst[i] = (bf16)src[i];
}

// ---------------- kernel 3: QKVG GEMM  [ROWS,DIM] x [DIM,NQ] + bias -> bf16 ----------------
__global__ __launch_bounds__(256)
void k_gemm_qkvg(const bf16* __restrict__ Xn, const bf16* __restrict__ W,
                 const float* __restrict__ bias, bf16* __restrict__ qkvg) {
    const int wave = (blockIdx.x * 256 + threadIdx.x) >> 5;   // 0..32767
    const int lane = threadIdx.x & 31;
    const int mt = wave >> 7;          // 256 row tiles
    const int nt = wave & 127;         // 128 col tiles
    const int mrow = lane & 15;
    const bf16* arow = Xn + (size_t)(mt * 16 + mrow) * DIM;

    v8f acc = zero8();
    for (int kc = 0; kc < DIM; kc += 32) {
        v16bf a = load_a32(arow, kc, lane);
        // B operand: lane = N column (lane&15), elements over K in A's table order
        U16B bu;
        for (int s = 0; s < 16; ++s)
            bu.e[s] = W[(size_t)(kc + (s >> 3) * 16 + ((lane >= 16) ? 8 : 0) + (s & 7)) * NQ + nt * 16 + mrow];
        v16bf b = bu.v;
        acc = wmma_bf16(a, b, acc);
    }
    const int hi8 = (lane >= 16) ? 8 : 0;
    const int n = nt * 16 + mrow;
    const float bv = bias[n];
    for (int r = 0; r < 8; ++r) {
        const int m = mt * 16 + r + hi8;
        qkvg[(size_t)m * NQ + n] = (bf16)(acc[r] + bv);
    }
}

// ---------------- kernel 4: Shepard attention per (b, h, 16-query tile) ----------------
__global__ __launch_bounds__(128)
void k_attn(const bf16* __restrict__ qkvg, float* __restrict__ Y) {
    const int wid  = threadIdx.x >> 5;
    const int lane = threadIdx.x & 31;
    const int gw = blockIdx.x * 4 + wid;           // 0..2047
    const int qt = gw & (QTILES - 1);
    const int bh = gw >> 7;
    const int h  = bh & (HEADS - 1);
    const int b  = bh >> 3;

    __shared__ bf16  kst[4][16 * DH];              // staged K tile (per wave)
    __shared__ float wst[4][16 * 32];              // weight tile 16x32 (per wave)
    bf16*  klds = kst[wid];
    float* wlds = wst[wid];

    const bf16* base = qkvg + (size_t)b * SEQ * NQ;
    const int qcol = h * DH;
    const int kcol = DQK + h * DH;
    const int vcol = 2 * DQK + h * DH;
    const int hi8  = (lane >= 16) ? 8 : 0;
    const int mrow = lane & 15;

    // wave-uniform LDS byte address of this wave's K-tile buffer (for TDM descriptor)
    const unsigned int klds_addr =
        (unsigned int)__builtin_amdgcn_readfirstlane((int)(unsigned int)(uintptr_t)klds);

    // Q tile in A-layout (dh=64 -> two K=32 chunks)
    const bf16* qrow = base + (size_t)(qt * 16 + mrow) * NQ + qcol;
    const v16bf qa0 = load_a32(qrow, 0, lane);
    const v16bf qa1 = load_a32(qrow, 32, lane);

    // |q|^2 per query row
    float q2p = 0.0f;
    {
        U16B u0, u1; u0.v = qa0; u1.v = qa1;
        for (int e = 0; e < 16; ++e) {
            float f0 = (float)u0.e[e], f1 = (float)u1.e[e];
            q2p += f0 * f0 + f1 * f1;
        }
    }
    q2p += __shfl_xor(q2p, 16, 32);                // lane L now holds q2[L&15]
    float q2m[8];
    for (int r = 0; r < 8; ++r) q2m[r] = __shfl(q2p, r + hi8, 32);

    v8f yacc[4];
    for (int t = 0; t < 4; ++t) yacc[t] = zero8();
    float rs[8];
    for (int r = 0; r < 8; ++r) rs[r] = 0.0f;

    for (int kt = 0; kt <= qt; kt += 2) {          // key tiles in pairs -> P is 16x32
        // prefetch next key-pair's K rows (global_prefetch_b8)
        if (kt + 2 <= qt) {
            __builtin_prefetch(base + (size_t)((kt + 2) * 16 + mrow) * NQ + kcol, 0, 1);
        }
        for (int st = 0; st < 2; ++st) {
            const int ktt = kt + st;
            // stage K tile (16 keys x 64 ch, 128B rows) into LDS via Tensor Data Mover
            {
                const unsigned long long ga =
                    (unsigned long long)(uintptr_t)(base + (size_t)(ktt * 16) * NQ + kcol);
                const unsigned int ga_lo =
                    (unsigned int)__builtin_amdgcn_readfirstlane((int)(unsigned int)ga);
                const unsigned int ga_hi =
                    (unsigned int)__builtin_amdgcn_readfirstlane((int)(unsigned int)(ga >> 32));
                tdm_load_k_tile(klds_addr, ga_lo, ga_hi);
                __builtin_amdgcn_s_wait_tensorcnt(0);
                asm volatile("" ::: "memory");     // keep LDS reads below the wait
            }
            // |k|^2 for this lane's key column (mrow)
            float k2v = 0.0f;
            {
                const v8bf* kp = (const v8bf*)(klds + mrow * DH);
                for (int i = 0; i < 8; ++i) {
                    v8bf t = kp[i];
                    for (int j = 0; j < 8; ++j) { float f = (float)t[j]; k2v += f * f; }
                }
            }
            // S = Q . K^T : B operand from LDS (lane = key mrow, elements = channels in table order)
            v8f s = zero8();
            for (int c2 = 0; c2 < 2; ++c2) {
                U16B bb;
                for (int n2 = 0; n2 < 16; ++n2) bb.e[n2] = klds[mrow * DH + c2 * 32 + (n2 >> 3) * 16 + hi8 + (n2 & 7)];
                s = wmma_bf16(c2 ? qa1 : qa0, bb.v, s);
            }
            // Shepard weights: w = 1/max(d2, eps^2), strictly-causal mask, row sums
            for (int r = 0; r < 8; ++r) {
                const int m_g = qt * 16 + r + hi8;
                const int n_g = ktt * 16 + mrow;
                const float d2 = q2m[r] + k2v - 2.0f * s[r];
                const float w  = (n_g >= m_g) ? 0.0f : (1.0f / fmaxf(d2, 1e-16f));
                rs[r] += w;
                wlds[(r + hi8) * 32 + st * 16 + mrow] = w;
            }
        }
        // P (16x32 bf16, A-layout) from LDS; Y += P @ V
        U16B p;
        for (int e = 0; e < 16; ++e) {
            const int key = (e >> 3) * 16 + hi8 + (e & 7);
            p.e[e] = (bf16)wlds[mrow * 32 + key];
        }
        const bf16* vrow = base + (size_t)(kt * 16) * NQ + vcol;  // lane = V column, elements = keys
        for (int t = 0; t < 4; ++t) {
            U16B vb;
            for (int e = 0; e < 16; ++e)
                vb.e[e] = vrow[(size_t)((e >> 3) * 16 + hi8 + (e & 7)) * NQ + t * 16 + mrow];
            yacc[t] = wmma_bf16(p.v, vb.v, yacc[t]);
        }
    }

    // normalize rows and store merged-head Y (f32)
    float inv[8];
    for (int r = 0; r < 8; ++r) {
        float t = rs[r];
        t += __shfl_xor(t, 1, 32);
        t += __shfl_xor(t, 2, 32);
        t += __shfl_xor(t, 4, 32);
        t += __shfl_xor(t, 8, 32);
        inv[r] = 1.0f / fmaxf(t, 1e-8f);
    }
    for (int t = 0; t < 4; ++t)
        for (int r = 0; r < 8; ++r) {
            const size_t m = (size_t)(b * SEQ + qt * 16 + r + hi8);
            Y[m * DVG + h * DH + t * 16 + mrow] = yacc[t][r] * inv[r];
        }
}

// ---------------- kernel 5: out = X + (Y .* G) @ W_out + b_out ----------------
__global__ __launch_bounds__(256)
void k_gemm_out(const float* __restrict__ Yin, const bf16* __restrict__ qkvg,
                const bf16* __restrict__ Wo, const float* __restrict__ X,
                const float* __restrict__ bout, float* __restrict__ Out) {
    const int wave = (blockIdx.x * 256 + threadIdx.x) >> 5;   // 0..16383
    const int lane = threadIdx.x & 31;
    const int mt = wave >> 6;          // 256 row tiles
    const int nt = wave & 63;          // 64 col tiles
    const int hi8  = (lane >= 16) ? 8 : 0;
    const int mrow = lane & 15;
    const int m = mt * 16 + mrow;
    const float* yrow = Yin + (size_t)m * DVG;
    const bf16*  grow = qkvg + (size_t)m * NQ + (2 * DQK + DVG);  // G columns

    v8f acc = zero8();
    for (int kc = 0; kc < DVG; kc += 32) {
        const int kb = kc + hi8;
        v8f  y0 = *(const v8f*)(yrow + kb);
        v8f  y1 = *(const v8f*)(yrow + kb + 16);
        v8bf g0 = *(const v8bf*)(grow + kb);
        v8bf g1 = *(const v8bf*)(grow + kb + 16);
        U16B a;
        for (int e = 0; e < 8; ++e) {
            a.e[e]     = (bf16)(y0[e] * (float)g0[e]);
            a.e[e + 8] = (bf16)(y1[e] * (float)g1[e]);
        }
        U16B bmat;
        for (int s = 0; s < 16; ++s)
            bmat.e[s] = Wo[(size_t)(kb + (s >> 3) * 16 + (s & 7)) * DIM + nt * 16 + mrow];
        acc = wmma_bf16(a.v, bmat.v, acc);
    }
    const int n = nt * 16 + mrow;
    const float bv = bout[n];
    for (int r = 0; r < 8; ++r) {
        const size_t mm = (size_t)(mt * 16 + r + hi8);
        Out[mm * DIM + n] = X[mm * DIM + n] + acc[r] + bv;
    }
}

// ---------------- launcher ----------------
extern "C" void kernel_launch(void* const* d_in, const int* in_sizes, int n_in,
                              void* d_out, int out_size, void* d_ws, size_t ws_size,
                              hipStream_t stream) {
    (void)in_sizes; (void)n_in; (void)out_size; (void)ws_size;
    const float* X     = (const float*)d_in[0];
    // d_in[1] = mask (bool) -- causality computed analytically, ignored
    const float* gamma = (const float*)d_in[2];
    const float* beta  = (const float*)d_in[3];
    const float* Wq    = (const float*)d_in[4];
    const float* bq    = (const float*)d_in[5];
    const float* Wo    = (const float*)d_in[6];
    const float* bo    = (const float*)d_in[7];
    float* Out = (float*)d_out;

    // workspace carve-up (bytes)
    char* ws = (char*)d_ws;
    bf16*  xn_bf   = (bf16*)(ws);                                   // ROWS*DIM*2   = 8 MiB
    bf16*  wq_bf   = (bf16*)(ws + (size_t)ROWS * DIM * 2);          // DIM*NQ*2     = 4 MiB
    bf16*  wo_bf   = (bf16*)(ws + (size_t)ROWS * DIM * 2
                                + (size_t)DIM * NQ * 2);            // DVG*DIM*2    = 1 MiB
    bf16*  qkvg_bf = (bf16*)(ws + (size_t)ROWS * DIM * 2
                                + (size_t)DIM * NQ * 2
                                + (size_t)DVG * DIM * 2);           // ROWS*NQ*2    = 16 MiB
    float* y_f32   = (float*)(ws + (size_t)ROWS * DIM * 2
                                 + (size_t)DIM * NQ * 2
                                 + (size_t)DVG * DIM * 2
                                 + (size_t)ROWS * NQ * 2);          // ROWS*DVG*4   = 8 MiB

    // 1) LayerNorm -> bf16
    k_layernorm<<<ROWS, 256, 0, stream>>>(X, gamma, beta, xn_bf);
    // 2) weight conversions
    {
        const int nq = DIM * NQ;
        k_cvt_bf16<<<(nq + 255) / 256, 256, 0, stream>>>(Wq, wq_bf, nq);
        const int no = DVG * DIM;
        k_cvt_bf16<<<(no + 255) / 256, 256, 0, stream>>>(Wo, wo_bf, no);
    }
    // 3) QKVG GEMM: (ROWS/16)*(NQ/16) = 32768 waves, 8 waves/block
    k_gemm_qkvg<<<32768 / 8, 256, 0, stream>>>(xn_bf, wq_bf, bq, qkvg_bf);
    // 4) attention: BATCH*HEADS*QTILES = 2048 waves, 4 waves/block
    k_attn<<<2048 / 4, 128, 0, stream>>>(qkvg_bf, y_f32);
    // 5) output projection + residual: (ROWS/16)*(DIM/16) = 16384 waves
    k_gemm_out<<<16384 / 8, 256, 0, stream>>>(y_f32, qkvg_bf, wo_bf, X, bo, Out);
}
